// DynamicAttention_90915867722022
// MI455X (gfx1250) — matrix operand order repaired and checked by reading, not compiled, not run on hardware
//
#include <hip/hip_runtime.h>

#define DIM   768
#define NHEAD 12
#define HDIM  64
#define BATCH 8
#define SEQ   1024
#define ROWS  (BATCH*SEQ)     // 8192
#define SCALE 0.125f          // 64^-0.5

typedef __bf16 bf16_t;
typedef __attribute__((ext_vector_type(16))) __bf16 v16bf;
typedef __attribute__((ext_vector_type(8)))  __bf16 v8bf;
typedef __attribute__((ext_vector_type(8)))  float  v8f;

#define WMMA_BF16(a, b, c) \
  __builtin_amdgcn_wmma_f32_16x16x32_bf16(false, (a), false, (b), (short)0, (c), false, false)

static __device__ __forceinline__ v16bf cat8(v8bf lo, v8bf hi) {
  return __builtin_shufflevector(lo, hi, 0,1,2,3,4,5,6,7,8,9,10,11,12,13,14,15);
}

// A-fragment: 16 rows x 32 k; lane ln = row, lo chunk k = 16*hf.., hi chunk k = 16*hf+8..
static __device__ __forceinline__ v16bf load_a(const bf16_t* row, int k0, int hf) {
  return cat8(*(const v8bf*)(row + k0 + 16 * hf),
              *(const v8bf*)(row + k0 + 16 * hf + 8));
}

// async copy 8 bytes global -> LDS (CDNA5 GLOBAL_LOAD_ASYNC_TO_LDS_B64, ASYNCcnt)
static __device__ __forceinline__ void async_copy_b64(const float* g, float* l) {
  unsigned lds = (unsigned)(uintptr_t)l;   // low 32 bits of generic addr == LDS offset
  asm volatile("global_load_async_to_lds_b64 %0, %1, off" :: "v"(lds), "v"(g) : "memory");
}
static __device__ __forceinline__ void wait_async_le1() {
  asm volatile("s_wait_asynccnt 0x1" ::: "memory");
}
static __device__ __forceinline__ void wait_async_0() {
  asm volatile("s_wait_asynccnt 0x0" ::: "memory");
}

// ---------------------------------------------------------------- convert
__global__ __launch_bounds__(256)
void cvt_f32_bf16_kernel(const float* __restrict__ src, unsigned short* __restrict__ dst, int n) {
  bf16_t* d = (bf16_t*)dst;
  int i = blockIdx.x * blockDim.x + threadIdx.x;
  int stride = gridDim.x * blockDim.x;
  for (; i < n; i += stride) d[i] = (bf16_t)src[i];
}

// ---------------------------------------------------------------- QKV GEMM
// out[i,j] = sum_k x[i,k] * w_qkv[j,k];  scatter into q/k/v^T bf16 buffers.
// 8 waves/block; each wave owns a 32x64 output tile (block = 256x64).
// Software-pipelined + fully unrolled: next-step fragments issued before
// current WMMAs; full unroll renames the rotation (no v_mov / hazard nops).
__global__ __launch_bounds__(256)
void qkv_gemm_kernel(const unsigned short* __restrict__ xb_, const unsigned short* __restrict__ wq_,
                     unsigned short* __restrict__ qb_, unsigned short* __restrict__ kb_,
                     unsigned short* __restrict__ vt_) {
  const bf16_t* xb = (const bf16_t*)xb_;
  const bf16_t* wq = (const bf16_t*)wq_;
  bf16_t* qb = (bf16_t*)qb_;
  bf16_t* kb = (bf16_t*)kb_;
  bf16_t* vt = (bf16_t*)vt_;

  const int lane = threadIdx.x & 31, w = threadIdx.x >> 5;
  const int ln = lane & 15, hf = lane >> 4;
  const int m0 = blockIdx.x * 256 + w * 32;
  const int n0 = blockIdx.y * 64;

  const bf16_t* arow0 = xb + (size_t)(m0 + ln) * DIM;
  const bf16_t* arow1 = xb + (size_t)(m0 + 16 + ln) * DIM;

  v8f acc0[4] = {}, acc1[4] = {};
  v16bf a0 = load_a(arow0, 0, hf);
  v16bf a1 = load_a(arow1, 0, hf);
  v16bf b[4];
#pragma unroll
  for (int n = 0; n < 4; ++n)
    b[n] = *(const v16bf*)(wq + (size_t)(n0 + n * 16 + ln) * DIM + 16 * hf);

#pragma unroll
  for (int kk = 0; kk < DIM; kk += 32) {
    v16bf na0 = a0, na1 = a1, nb[4] = {b[0], b[1], b[2], b[3]};
    if (kk + 32 < DIM) {
      na0 = load_a(arow0, kk + 32, hf);
      na1 = load_a(arow1, kk + 32, hf);
#pragma unroll
      for (int n = 0; n < 4; ++n)
        nb[n] = *(const v16bf*)(wq + (size_t)(n0 + n * 16 + ln) * DIM + kk + 32 + 16 * hf);
    }
#pragma unroll
    for (int n = 0; n < 4; ++n) {
      acc0[n] = WMMA_BF16(a0, b[n], acc0[n]);
      acc1[n] = WMMA_BF16(a1, b[n], acc1[n]);
    }
    a0 = na0; a1 = na1;
#pragma unroll
    for (int n = 0; n < 4; ++n) b[n] = nb[n];
  }

#pragma unroll
  for (int n = 0; n < 4; ++n) {
    const int j0 = n0 + n * 16;       // tile-uniform: which of q/k/v
    const int sel = j0 / DIM;
    const int c = (j0 % DIM) + ln;
    const int hh = c >> 6, d = c & 63;
#pragma unroll
    for (int half = 0; half < 2; ++half) {
#pragma unroll
      for (int r = 0; r < 8; ++r) {
        const int i = m0 + half * 16 + r + 8 * hf;
        const int bb = i >> 10, nn = i & 1023;
        bf16_t val = (bf16_t)(half ? acc1[n][r] : acc0[n][r]);
        if (sel == 0)      qb[(((size_t)bb * NHEAD + hh) * SEQ + nn) * HDIM + d] = val;
        else if (sel == 1) kb[(((size_t)bb * NHEAD + hh) * SEQ + nn) * HDIM + d] = val;
        else               vt[(((size_t)bb * NHEAD + hh) * HDIM + d) * SEQ + nn] = val;
      }
    }
  }
}

// ---------------------------------------------------------------- attention
// grid = (NHEAD, SEQ/16); block = 256 = 8 waves; wave w handles batch b = w
// for the same (h, q-tile): sigmoid(mask) tile shared 8x, async-prefetched.
__global__ __launch_bounds__(256)
void attn_kernel(const unsigned short* __restrict__ qb_, const unsigned short* __restrict__ kb_,
                 const unsigned short* __restrict__ vt_, const float* __restrict__ mask,
                 unsigned short* __restrict__ ao_) {
  const bf16_t* qb = (const bf16_t*)qb_;
  const bf16_t* kb = (const bf16_t*)kb_;
  const bf16_t* vt = (const bf16_t*)vt_;
  bf16_t* ao = (bf16_t*)ao_;

  __shared__ float  s_raw[2][16 * 32];  // raw mask tiles, async double-buffered
  __shared__ float  s_g[2][16 * 32];    // SCALE * sigmoid(mask)
  __shared__ __bf16 s_p[8 * 16 * 32];   // per-wave P staging (C-layout -> A-layout)

  const int tid = threadIdx.x;
  const int lane = tid & 31, b = tid >> 5;     // wave index == batch index
  const int ln = lane & 15, hf = lane >> 4;
  const int h = blockIdx.x, q0 = blockIdx.y * 16;
  __bf16* pw = s_p + b * 512;

  const bf16_t* qbase = qb + (((size_t)b * NHEAD + h) * SEQ + q0) * HDIM;
  const bf16_t* kbase = kb + ((size_t)b * NHEAD + h) * SEQ * HDIM;
  const bf16_t* vbase = vt + ((size_t)b * NHEAD + h) * HDIM * SEQ;
  const float*  mbase = mask + ((size_t)h * SEQ + q0) * SEQ;

  // this thread's 2-element slice of the 16x32 mask tile
  const int e0 = tid * 2;
  const int mrow_i = e0 >> 5, mcol_i = e0 & 31;

  // Q fragments (16 rows x 64 k -> two 16x32 A-frags)
  const bf16_t* qrow = qbase + (size_t)ln * HDIM;
  v16bf aq0 = load_a(qrow, 0, hf);
  v16bf aq1 = load_a(qrow, 32, hf);

  v8f o[4] = {};
  float mrow[8], lrow[8];
#pragma unroll
  for (int r = 0; r < 8; ++r) { mrow[r] = -3.0e38f; lrow[r] = 0.f; }

  // prologue: async-prefetch mask tile 0
  async_copy_b64(&mbase[(size_t)mrow_i * SEQ + mcol_i], &s_raw[0][e0]);

  const int nsteps = SEQ / 32;
  for (int s = 0; s < nsteps; ++s) {
    const int m0 = s * 32;
    const int buf = s & 1;

    // async-prefetch tile s+1 while tile s is consumed below
    if (s + 1 < nsteps) {
      async_copy_b64(&mbase[(size_t)mrow_i * SEQ + m0 + 32 + mcol_i], &s_raw[buf ^ 1][e0]);
      wait_async_le1();                 // tile s (oldest) landed in LDS
    } else {
      wait_async_0();
    }
    __syncthreads();                    // all waves' slices of raw[buf] visible

    // gating tile: g = SCALE * sigmoid(raw)
    s_g[buf][e0]     = SCALE / (1.f + __expf(-s_raw[buf][e0]));
    s_g[buf][e0 + 1] = SCALE / (1.f + __expf(-s_raw[buf][e0 + 1]));
    __syncthreads();

    // S = Q K^T for 32 keys (two 16x16 tiles, K-dim 64 = 2 WMMA steps each)
    const bf16_t* krow = kbase + (size_t)(m0 + ln) * HDIM;
    v8f s0 = {}, s1 = {};
    s0 = WMMA_BF16(aq0, *(const v16bf*)(krow + 16 * hf), s0);
    s0 = WMMA_BF16(aq1, *(const v16bf*)(krow + 32 + 16 * hf), s0);
    s1 = WMMA_BF16(aq0, *(const v16bf*)(krow + 16 * HDIM + 16 * hf), s1);
    s1 = WMMA_BF16(aq1, *(const v16bf*)(krow + 16 * HDIM + 32 + 16 * hf), s1);

    // gated online softmax; rows live across 16-lane halves in the C layout
#pragma unroll
    for (int r = 0; r < 8; ++r) {
      const int row = r + 8 * hf;
      float l0 = s0[r] * s_g[buf][row * 32 + ln];
      float l1 = s1[r] * s_g[buf][row * 32 + 16 + ln];
      float rmax = fmaxf(l0, l1);
#pragma unroll
      for (int off = 1; off < 16; off <<= 1) rmax = fmaxf(rmax, __shfl_xor(rmax, off, 32));
      const float mnew = fmaxf(mrow[r], rmax);
      const float corr = __expf(mrow[r] - mnew);
      const float p0 = __expf(l0 - mnew);
      const float p1 = __expf(l1 - mnew);
      float psum = p0 + p1;
#pragma unroll
      for (int off = 1; off < 16; off <<= 1) psum += __shfl_xor(psum, off, 32);
      lrow[r] = lrow[r] * corr + psum;
      mrow[r] = mnew;
#pragma unroll
      for (int n = 0; n < 4; ++n) o[n][r] *= corr;
      pw[row * 32 + ln]      = (__bf16)p0;
      pw[row * 32 + 16 + ln] = (__bf16)p1;
    }

    // reload P as one 16x32 A fragment from this wave's private LDS region
    v16bf ap = cat8(*(const v8bf*)(pw + ln * 32 + 16 * hf),
                    *(const v8bf*)(pw + ln * 32 + 16 * hf + 8));
    // O += P V  (V^T layout -> contiguous 32B B-frag rows)
#pragma unroll
    for (int n = 0; n < 4; ++n) {
      v16bf bv = *(const v16bf*)(vbase + (size_t)(n * 16 + ln) * SEQ + m0 + 16 * hf);
      o[n] = WMMA_BF16(ap, bv, o[n]);
    }
  }

  // normalize and write to [B, N, H*64] bf16 for the projection GEMM
  bf16_t* obase = ao + ((size_t)b * SEQ + q0) * DIM + h * HDIM;
#pragma unroll
  for (int r = 0; r < 8; ++r) {
    const int row = r + 8 * hf;
    const float inv = 1.f / lrow[r];
#pragma unroll
    for (int n = 0; n < 4; ++n)
      obase[(size_t)row * DIM + n * 16 + ln] = (bf16_t)(o[n][r] * inv);
  }
}

// ---------------------------------------------------------------- proj GEMM
__global__ __launch_bounds__(256)
void proj_gemm_kernel(const unsigned short* __restrict__ ao_, const unsigned short* __restrict__ wp_,
                      const float* __restrict__ bias, float* __restrict__ out) {
  const bf16_t* ao = (const bf16_t*)ao_;
  const bf16_t* wp = (const bf16_t*)wp_;

  const int lane = threadIdx.x & 31, w = threadIdx.x >> 5;
  const int ln = lane & 15, hf = lane >> 4;
  const int m0 = blockIdx.x * 256 + w * 32;
  const int n0 = blockIdx.y * 64;

  const bf16_t* arow0 = ao + (size_t)(m0 + ln) * DIM;
  const bf16_t* arow1 = ao + (size_t)(m0 + 16 + ln) * DIM;

  v8f acc0[4] = {}, acc1[4] = {};
  v16bf a0 = load_a(arow0, 0, hf);
  v16bf a1 = load_a(arow1, 0, hf);
  v16bf b[4];
#pragma unroll
  for (int n = 0; n < 4; ++n)
    b[n] = *(const v16bf*)(wp + (size_t)(n0 + n * 16 + ln) * DIM + 16 * hf);

#pragma unroll
  for (int kk = 0; kk < DIM; kk += 32) {
    v16bf na0 = a0, na1 = a1, nb[4] = {b[0], b[1], b[2], b[3]};
    if (kk + 32 < DIM) {
      na0 = load_a(arow0, kk + 32, hf);
      na1 = load_a(arow1, kk + 32, hf);
#pragma unroll
      for (int n = 0; n < 4; ++n)
        nb[n] = *(const v16bf*)(wp + (size_t)(n0 + n * 16 + ln) * DIM + kk + 32 + 16 * hf);
    }
#pragma unroll
    for (int n = 0; n < 4; ++n) {
      acc0[n] = WMMA_BF16(a0, b[n], acc0[n]);
      acc1[n] = WMMA_BF16(a1, b[n], acc1[n]);
    }
    a0 = na0; a1 = na1;
#pragma unroll
    for (int n = 0; n < 4; ++n) b[n] = nb[n];
  }

#pragma unroll
  for (int n = 0; n < 4; ++n) {
    const int j = n0 + n * 16 + ln;
    const float bj = bias[j];
#pragma unroll
    for (int r = 0; r < 8; ++r) {
      out[(size_t)(m0 + r + 8 * hf) * DIM + j]      = acc0[n][r] + bj;
      out[(size_t)(m0 + 16 + r + 8 * hf) * DIM + j] = acc1[n][r] + bj;
    }
  }
}

// ---------------------------------------------------------------- host
extern "C" void kernel_launch(void* const* d_in, const int* in_sizes, int n_in,
                              void* d_out, int out_size, void* d_ws, size_t ws_size,
                              hipStream_t stream) {
  (void)in_sizes; (void)n_in; (void)out_size; (void)ws_size;
  const float* x        = (const float*)d_in[0];
  const float* w_qkv    = (const float*)d_in[1];
  const float* w_proj   = (const float*)d_in[2];
  const float* b_proj   = (const float*)d_in[3];
  const float* att_mask = (const float*)d_in[4];
  float* out = (float*)d_out;

  char* ws = (char*)d_ws;
  size_t off = 0;
  auto alloc = [&](size_t bytes) -> void* {
    void* p = ws + off;
    off += (bytes + 255) & ~(size_t)255;
    return p;
  };
  unsigned short* xb  = (unsigned short*)alloc((size_t)ROWS * DIM * 2);
  unsigned short* wqb = (unsigned short*)alloc((size_t)3 * DIM * DIM * 2);
  unsigned short* wpb = (unsigned short*)alloc((size_t)DIM * DIM * 2);
  unsigned short* qb  = (unsigned short*)alloc((size_t)ROWS * DIM * 2);
  unsigned short* kb  = (unsigned short*)alloc((size_t)ROWS * DIM * 2);
  unsigned short* vtb = (unsigned short*)alloc((size_t)ROWS * DIM * 2);
  unsigned short* aob = (unsigned short*)alloc((size_t)ROWS * DIM * 2);

  cvt_f32_bf16_kernel<<<512, 256, 0, stream>>>(x, xb, ROWS * DIM);
  cvt_f32_bf16_kernel<<<512, 256, 0, stream>>>(w_qkv, wqb, 3 * DIM * DIM);
  cvt_f32_bf16_kernel<<<256, 256, 0, stream>>>(w_proj, wpb, DIM * DIM);

  qkv_gemm_kernel<<<dim3(ROWS / 256, 3 * DIM / 64), 256, 0, stream>>>(xb, wqb, qb, kb, vtb);
  attn_kernel<<<dim3(NHEAD, SEQ / 16), 256, 0, stream>>>(qb, kb, vtb, att_mask, aob);
  proj_gemm_kernel<<<dim3(ROWS / 256, DIM / 64), 256, 0, stream>>>(aob, wpb, b_proj, out);
}
